// McG_LSTM_8581344658101
// MI455X (gfx1250) — matrix operand order repaired and checked by reading, not compiled, not run on hardware
//
#include <hip/hip_runtime.h>
#include <hip/hip_bf16.h>

// ---------------------------------------------------------------------------
// 4-layer LSTM (hidden 25, batch 4096, len 512), persistent-wave design.
//   - each wave owns 16 batch rows for the whole 512-step sequence
//   - gates[16x112] = [x|h](16x64,f16) @ W^T(64x112,f16): 14 wmma/layer/step
//   - ALL weight B-fragments live in registers (448 VGPRs; occupancy is
//     parallelism-capped at ~1 wave/SIMD, so the 1024-VGPR file is free)
//   - h written back in A-fragment layout (f16) each step; c/h f32 in LDS
// ---------------------------------------------------------------------------

typedef __attribute__((ext_vector_type(16))) _Float16 v16h;
typedef __attribute__((ext_vector_type(8)))  float    v8f;

#define NW     2              // waves per block
#define TPB    (NW * 32)
#define BATCH  4096
#define SEQ    512
#define NGT    112            // 100 gates padded to 7 tiles of 16

__device__ __forceinline__ float sig_f(float x) {
    // 1/(1+2^(-x*log2e)) -> v_exp_f32 + v_rcp_f32 (trans pipe, co-executes with WMMA)
    return __builtin_amdgcn_rcpf(1.0f + __builtin_amdgcn_exp2f(-1.4426950408889634f * x));
}
__device__ __forceinline__ float tanh_f(float x) {
    return 2.0f * sig_f(2.0f * x) - 1.0f;
}

// A-fragment slot for K=j (16-bit A 16x32 layout, ISA 7.12.2):
//  lanes 0-15: halves 0-7 -> K 0-7,  halves 8-15 -> K 16-23
//  lanes16-31: halves 0-7 -> K 8-15, halves 8-15 -> K 24-31
__device__ __forceinline__ int afrag_lane_hi(int j) { return (j >> 3) & 1; }
__device__ __forceinline__ int afrag_half(int j)    { return (j & 7) | ((j >> 4) << 3); }

__global__ __launch_bounds__(TPB)
void McG_LSTM_8581344658101_kernel(
    const float* __restrict__ x,
    const float* __restrict__ Wih0, const float* __restrict__ Whh0,
    const float* __restrict__ bih0, const float* __restrict__ bhh0,
    const float* __restrict__ Wih1, const float* __restrict__ Whh1,
    const float* __restrict__ bih1, const float* __restrict__ bhh1,
    const float* __restrict__ Wih2, const float* __restrict__ Whh2,
    const float* __restrict__ bih2, const float* __restrict__ bhh2,
    const float* __restrict__ Wih3, const float* __restrict__ Whh3,
    const float* __restrict__ bih3, const float* __restrict__ bhh3,
    const float* __restrict__ Wfc,  const float* __restrict__ bfc,
    float* __restrict__ out)
{
    extern __shared__ __align__(32) char smem[];

    // ---- LDS layout -------------------------------------------------------
    // fragB : half[4][7][2][32][16]  = 57344 B  (setup staging only)
    // bias  : float[4][112]          =  1792 B
    // wfc   : float[26]              ->  128 B  ([25] = b_fc)
    // per-wave (x NW):
    //   fragA: half[4][32][16]  (4096 B)  h-state in A-fragment layout (f16)
    //   cbuf : float[4][400]    (6400 B)  idx = j*16+b
    //   hbuf : float[4][400]    (6400 B)
    //   gates: float[112][16]   (7168 B)  [gate][batchRow]
    _Float16* fragB = (_Float16*)smem;                          // 57344
    float*    biasL = (float*)(smem + 57344);                   //  1792
    float*    wfcL  = (float*)(smem + 57344 + 1792);            //   128
    char*     pw    = smem + 59264;
    _Float16* fragA = (_Float16*)pw;                            // NW*4096
    float*    cbuf  = (float*)(pw + NW * 4096);                 // NW*6400
    float*    hbuf  = (float*)(pw + NW * 4096 + NW * 6400);     // NW*6400
    float*    gbuf  = (float*)(pw + NW * 4096 + NW * 12800);    // NW*7168

    const int tid = threadIdx.x;
    const float* WihA[4] = {Wih0, Wih1, Wih2, Wih3};
    const float* WhhA[4] = {Whh0, Whh1, Whh2, Whh3};
    const float* bihA[4] = {bih0, bih1, bih2, bih3};
    const float* bhhA[4] = {bhh0, bhh1, bhh2, bhh3};

    // ---- one-time setup: swizzle weights into B fragments, zero state -----
    for (int i = tid; i < NW * 4 * 512; i += TPB) fragA[i] = (_Float16)0.0f;
    for (int i = tid; i < NW * 4 * 400; i += TPB) {
        cbuf[i] = 0.0f; hbuf[i] = 0.0f;
    }
    // B fragment (K x N, 16-bit): lanes 0-15 hold K 0-7 (halves 0-7), K 16-23 (halves 8-15)
    // for column N=lane&15; lanes 16-31 hold K 8-15, K 24-31 (matches A-fragment layout).
    for (int e = tid; e < 4 * 7 * 2 * 512; e += TPB) {
        int l  = e / 7168;
        int r  = e % 7168;
        int t  = r / 1024;
        int c  = (r % 1024) / 512;
        int q  = r % 512;
        int ln = q >> 4;
        int hh = q & 15;
        int n  = ln & 15;
        int k  = (hh & 7) + 8 * (ln >> 4) + 16 * (hh >> 3);
        int g  = 16 * t + n;
        float v = 0.0f;
        if (g < 100) {
            if (c == 0) { int ind = (l == 0) ? 1 : 25; if (k < ind) v = WihA[l][g * ind + k]; }
            else        {                              if (k < 25)  v = WhhA[l][g * 25  + k]; }
        }
        fragB[e] = (_Float16)v;
    }
    for (int i = tid; i < 4 * NGT; i += TPB) {
        int l = i / NGT, g = i % NGT;
        biasL[i] = (g < 100) ? (bihA[l][g] + bhhA[l][g]) : 0.0f;
    }
    if (tid < 25) wfcL[tid] = Wfc[tid];
    if (tid == 25) wfcL[25] = bfc[0];
    __syncthreads();

    // ---- per-wave persistent state ----------------------------------------
    const int wv     = tid >> 5;
    const int lane   = tid & 31;
    const int nl     = lane & 15;
    const int hi     = lane >> 4;
    const int batch0 = (blockIdx.x * NW + wv) * 16;

    _Float16* myA = fragA + wv * 4 * 512;
    float*    myC = cbuf  + wv * 4 * 400;
    float*    myH = hbuf  + wv * 4 * 400;
    float*    myG = gbuf  + wv * NGT * 16;

    // Hoist ALL loop-invariant operands into registers:
    //  - 56 weight B-fragments (4 layers x 7 tiles x 2 K-chunks) = 448 VGPRs
    //  - 28 per-lane biases, 26 fc weights
    v16h breg[56];
#pragma unroll
    for (int i = 0; i < 56; ++i)
        breg[i] = *(const v16h*)(fragB + i * 512 + lane * 16);

    float biasr[28];
#pragma unroll
    for (int l = 0; l < 4; ++l)
#pragma unroll
        for (int t = 0; t < 7; ++t)
            biasr[l * 7 + t] = biasL[l * NGT + 16 * t + nl];
    float wreg[26];
#pragma unroll
    for (int j = 0; j < 26; ++j) wreg[j] = wfcL[j];

    // ---- sequence loop ----------------------------------------------------
    for (int t = 0; t < SEQ; ++t) {
        float xv = x[(size_t)t * BATCH + batch0 + nl];
        if (t + 1 < SEQ)
            __builtin_prefetch(x + (size_t)(t + 1) * BATCH + batch0, 0, 0);

        // layer-1 input fragment: K=0 column = x_t  (slot: lanes 0-15, half 0)
        v16h ax;
#pragma unroll
        for (int i = 0; i < 16; ++i) ax[i] = (_Float16)0.0f;
        ax[0] = (hi == 0) ? (_Float16)xv : (_Float16)0.0f;

#pragma unroll
        for (int l = 0; l < 4; ++l) {
            // A operands: chunk0 = layer input, chunk1 = own h (prev step)
            v16h a0 = (l == 0) ? ax
                               : *(const v16h*)(myA + (l - 1) * 512 + lane * 16);
            v16h a1 = *(const v16h*)(myA + l * 512 + lane * 16);

#pragma unroll
            for (int tt = 0; tt < 7; ++tt) {
                float bv = biasr[l * 7 + tt];
                v8f acc;
#pragma unroll
                for (int r = 0; r < 8; ++r) acc[r] = bv;
                acc = __builtin_amdgcn_wmma_f32_16x16x32_f16(
                          false, a0, false, breg[(l * 7 + tt) * 2 + 0],
                          (short)0, acc, false, false);
                acc = __builtin_amdgcn_wmma_f32_16x16x32_f16(
                          false, a1, false, breg[(l * 7 + tt) * 2 + 1],
                          (short)0, acc, false, false);
                // D layout: lane = N + 16*(M>=8), VGPR r = M&7 -> contiguous
                // 8 rows for gate column g = 16*tt + nl
                *(v8f*)(myG + (16 * tt + nl) * 16 + 8 * hi) = acc;
            }
            asm volatile("s_wait_dscnt 0" ::: "memory");

            // elementwise LSTM update; write h back in A-fragment layout (f16)
            for (int idx = lane; idx < 400; idx += 32) {
                int b = idx & 15, j = idx >> 4;        // idx = j*16 + b
                float ii = myG[(j     ) * 16 + b];
                float ff = myG[(25 + j) * 16 + b];
                float gg = myG[(50 + j) * 16 + b];
                float oo = myG[(75 + j) * 16 + b];
                float c0 = myC[l * 400 + idx];
                float i_ = sig_f(ii), f_ = sig_f(ff);
                float g_ = tanh_f(gg), o_ = sig_f(oo);
                float cn = f_ * c0 + i_ * g_;
                float hn = o_ * tanh_f(cn);
                myC[l * 400 + idx] = cn;
                myH[l * 400 + idx] = hn;
                int dl = b + 16 * afrag_lane_hi(j);
                int dh = afrag_half(j);
                myA[l * 512 + dl * 16 + dh] = (_Float16)hn;
            }
            asm volatile("s_wait_dscnt 0" ::: "memory");
        }

        // y_t = h4 . Wfc + b_fc  (one batch row per low lane)
        if (hi == 0) {
            float s = wreg[25];
#pragma unroll
            for (int j = 0; j < 25; ++j)
                s += myH[3 * 400 + j * 16 + nl] * wreg[j];
            out[(size_t)t * BATCH + batch0 + nl] = s;
        }
    }

    // ---- final states: h1,c1,h2,c2,h3,c3,h4,c4 each [4096,25] -------------
    asm volatile("s_wait_dscnt 0" ::: "memory");
    const size_t YOFF = (size_t)SEQ * BATCH;
#pragma unroll
    for (int l = 0; l < 4; ++l) {
        for (int idx = lane; idx < 400; idx += 32) {
            int b = idx & 15, j = idx >> 4;
            size_t row = (size_t)(batch0 + b) * 25 + j;
            out[YOFF + (size_t)(2 * l)     * (BATCH * 25) + row] = myH[l * 400 + idx];
            out[YOFF + (size_t)(2 * l + 1) * (BATCH * 25) + row] = myC[l * 400 + idx];
        }
    }
}

extern "C" void kernel_launch(void* const* d_in, const int* in_sizes, int n_in,
                              void* d_out, int out_size, void* d_ws, size_t ws_size,
                              hipStream_t stream) {
    (void)in_sizes; (void)n_in; (void)d_ws; (void)ws_size; (void)out_size;
    const float* x    = (const float*)d_in[0];
    const float* W[16];
    for (int i = 0; i < 16; ++i) W[i] = (const float*)d_in[1 + i];
    const float* Wfc  = (const float*)d_in[17];
    const float* bfcp = (const float*)d_in[18];
    float* out = (float*)d_out;

    // LDS: 59264 (shared weights/bias) + NW*24064 (per-wave state) = 107392 B
    const size_t smemBytes = 59264 + (size_t)NW * (4096 + 6400 + 6400 + 7168);
    const int grid = BATCH / (16 * NW);   // 128 blocks x 2 waves = 256 wave-tiles

    McG_LSTM_8581344658101_kernel<<<grid, TPB, smemBytes, stream>>>(
        x,
        W[0],  W[1],  W[2],  W[3],
        W[4],  W[5],  W[6],  W[7],
        W[8],  W[9],  W[10], W[11],
        W[12], W[13], W[14], W[15],
        Wfc, bfcp, out);
}
